// GCN_41979010351255
// MI455X (gfx1250) — matrix operand order repaired and checked by reading, not compiled, not run on hardware
//
#include <hip/hip_runtime.h>
#include <hip/hip_bf16.h>

// ---- problem constants (match reference) ----
#define GN   50000   // nodes
#define GE   800000  // edges
#define GFIN 128     // input features
#define GH   64      // hidden
#define GC   10      // classes
#define GB   128     // graphs

typedef __attribute__((ext_vector_type(16))) _Float16 v16h;
typedef __attribute__((ext_vector_type(8)))  _Float16 v8h;
typedef __attribute__((ext_vector_type(8)))  float    v8f;

// ---------------- degree / normalization ----------------
__global__ void GCN_fill_deg(float* __restrict__ deg) {
    int i = blockIdx.x * blockDim.x + threadIdx.x;
    if (i < GN) deg[i] = 1.0f;               // self-loop contribution
}

__global__ void GCN_deg_scatter(const int* __restrict__ dst, float* __restrict__ deg) {
    int e = blockIdx.x * blockDim.x + threadIdx.x;
    if (e < GE) atomicAdd(&deg[dst[e]], 1.0f);
}

__global__ void GCN_rsqrt(const float* __restrict__ deg, float* __restrict__ dis) {
    int i = blockIdx.x * blockDim.x + threadIdx.x;
    if (i < GN) dis[i] = rsqrtf(deg[i]);     // deg >= 1 always
}

// ---------------- f16 conversions ----------------
__global__ void GCN_cvt_f16(const float* __restrict__ in, _Float16* __restrict__ out, int n) {
    int i = blockIdx.x * blockDim.x + threadIdx.x;
    if (i < n) out[i] = (_Float16)in[i];
}

// W is K x 64 row-major (f32); produce Wt = 64 x K row-major (f16)
__global__ void GCN_cvt_wt(const float* __restrict__ W, _Float16* __restrict__ Wt, int K) {
    int idx = blockIdx.x * blockDim.x + threadIdx.x;
    if (idx < 64 * K) {
        int n = idx / K, k = idx % K;
        Wt[n * K + k] = (_Float16)W[k * 64 + n];
    }
}

// ---------------- WMMA GEMM: out[N x 64] = Ah[N x K] * Wt^T ----------------
// One wave -> one 16x16 output tile. blockDim = 128 (4 waves = 4 col tiles = 64 cols).
// gridDim.x = N/16 row tiles. Register layouts used here (contiguous k, not the ISA table order):
//   A 16x32 f16:  lane m(+16*half) holds row m; elems 0..15 = K(16*half + 0..15)
//   B 32x16 f16:  lane %16 = col n; elems 0..15 = K(16*half + 0..15)  (contiguous in Wt row)
//   C/D f32:      lane %16 = col n; acc[r] = row (8*half + r)
template <int K>
__global__ void GCN_gemm_nx64(const _Float16* __restrict__ Ah,
                              const _Float16* __restrict__ Bt,
                              float* __restrict__ out) {
    const int lane = threadIdx.x & 31;
    const int wave = threadIdx.x >> 5;          // col tile 0..3
    const int m    = lane & 15;
    const int half = lane >> 4;
    const int row0 = blockIdx.x * 16;
    const int col  = wave * 16 + m;

    const _Float16* arow = Ah + (size_t)(row0 + m) * K;
    const _Float16* brow = Bt + (size_t)col * K;

    v8f acc = {};
#pragma unroll
    for (int kc = 0; kc < K; kc += 32) {
        v16h a = *(const v16h*)(arow + kc + 16 * half);
        v16h b = *(const v16h*)(brow + kc + 16 * half);
        acc = __builtin_amdgcn_wmma_f32_16x16x32_f16(false, a, false, b,
                                                     (short)0, acc, false, false);
    }
#pragma unroll
    for (int r = 0; r < 8; ++r)
        out[(size_t)(row0 + 8 * half + r) * 64 + col] = acc[r];
}

// ---------------- aggregation ----------------
// agg[i,f] = t[i,f]*dis[i]^2 + b[f]   (self-loop message + bias)
__global__ void GCN_agg_init(const float* __restrict__ t, const float* __restrict__ dis,
                             const float* __restrict__ bias, float* __restrict__ agg) {
    int idx = blockIdx.x * blockDim.x + threadIdx.x;
    if (idx < GN * GH) {
        int i = idx >> 6, f = idx & 63;
        float ds = dis[i];
        agg[idx] = t[idx] * ds * ds + bias[f];
    }
}

// one wave per edge, 2 features per lane; float atomics into agg[dst]
__global__ void GCN_edge_scatter(const float* __restrict__ t,
                                 const int* __restrict__ src, const int* __restrict__ dst,
                                 const float* __restrict__ dis, float* __restrict__ agg) {
    int e = blockIdx.x * 8 + (threadIdx.x >> 5);
    if (e >= GE) return;
    int lane = threadIdx.x & 31;
    int s = src[e], d = dst[e];
    float nrm = dis[s] * dis[d];
    float2 v = *(const float2*)(t + (size_t)s * 64 + lane * 2);
    atomicAdd(&agg[(size_t)d * 64 + lane * 2],     v.x * nrm);
    atomicAdd(&agg[(size_t)d * 64 + lane * 2 + 1], v.y * nrm);
}

__global__ void GCN_relu_cvt(const float* __restrict__ agg, _Float16* __restrict__ hh) {
    int idx = blockIdx.x * blockDim.x + threadIdx.x;
    if (idx < GN * GH) hh[idx] = (_Float16)fmaxf(agg[idx], 0.0f);
}

// ---------------- pooling + classifier ----------------
__global__ void GCN_zero_pool(float* __restrict__ sums, float* __restrict__ cnt) {
    int idx = blockIdx.x * blockDim.x + threadIdx.x;
    if (idx < GB * GH) sums[idx] = 0.0f;
    if (idx < GB) cnt[idx] = 0.0f;
}

__global__ void GCN_pool(const float* __restrict__ agg, const int* __restrict__ batch,
                         float* __restrict__ sums, float* __restrict__ cnt) {
    int idx = blockIdx.x * blockDim.x + threadIdx.x;
    if (idx < GN * GH) {
        int i = idx >> 6, f = idx & 63;
        int g = batch[i];
        atomicAdd(&sums[g * GH + f], agg[idx]);
        if (f == 0) atomicAdd(&cnt[g], 1.0f);
    }
}

__global__ void GCN_logits(const float* __restrict__ sums, const float* __restrict__ cnt,
                           const float* __restrict__ Wl, const float* __restrict__ bl,
                           float* __restrict__ out) {
    int idx = blockIdx.x * blockDim.x + threadIdx.x;
    if (idx >= GB * GC) return;
    int g = idx / GC, c = idx % GC;
    float inv = 1.0f / fmaxf(cnt[g], 1.0f);
    float acc = bl[c];
#pragma unroll
    for (int k = 0; k < GH; ++k)
        acc += sums[g * GH + k] * inv * Wl[k * GC + c];
    out[idx] = acc;
}

// ---------------- launch ----------------
extern "C" void kernel_launch(void* const* d_in, const int* in_sizes, int n_in,
                              void* d_out, int out_size, void* d_ws, size_t ws_size,
                              hipStream_t stream) {
    (void)in_sizes; (void)n_in; (void)out_size; (void)ws_size;

    const float* x    = (const float*)d_in[0];
    const int*   eidx = (const int*)  d_in[1];
    const int*   batch= (const int*)  d_in[2];
    const float* W1   = (const float*)d_in[3];
    const float* b1   = (const float*)d_in[4];
    const float* W2   = (const float*)d_in[5];
    const float* b2   = (const float*)d_in[6];
    const float* W3   = (const float*)d_in[7];
    const float* b3   = (const float*)d_in[8];
    const float* Wl   = (const float*)d_in[9];
    const float* bl   = (const float*)d_in[10];
    float* out        = (float*)d_out;

    const int* src = eidx;
    const int* dst = eidx + GE;

    // workspace carve-out (256B aligned), total ~38.9 MB
    char* p = (char*)d_ws;
    auto take = [&](size_t bytes) -> char* {
        char* r = p; p += (bytes + 255) & ~(size_t)255; return r;
    };
    float*    deg  = (float*)   take((size_t)GN * 4);
    float*    dis  = (float*)   take((size_t)GN * 4);
    _Float16* Wt1  = (_Float16*)take((size_t)64 * GFIN * 2);
    _Float16* Wt2  = (_Float16*)take((size_t)64 * GH * 2);
    _Float16* Wt3  = (_Float16*)take((size_t)64 * GH * 2);
    _Float16* Xh   = (_Float16*)take((size_t)GN * GFIN * 2); // reused as hh after GEMM1
    float*    t    = (float*)   take((size_t)GN * GH * 4);
    float*    agg  = (float*)   take((size_t)GN * GH * 4);
    float*    sums = (float*)   take((size_t)GB * GH * 4);
    float*    cnt  = (float*)   take((size_t)GB * 4);
    _Float16* hh   = Xh;

    const int T = 256;
    const int gN   = (GN + T - 1) / T;
    const int gE   = (GE + T - 1) / T;
    const int gNH  = (GN * GH + T - 1) / T;
    const int gNF  = (GN * GFIN + T - 1) / T;
    const int gEdg = (GE + 7) / 8;          // 8 edges per 256-thread block
    const int gTiles = GN / 16;             // 3125 row tiles (N divisible by 16)

    // normalization
    GCN_fill_deg   <<<gN, T, 0, stream>>>(deg);
    GCN_deg_scatter<<<gE, T, 0, stream>>>(dst, deg);
    GCN_rsqrt      <<<gN, T, 0, stream>>>(deg, dis);

    // weight / input conversion
    GCN_cvt_wt <<<(64 * GFIN + T - 1) / T, T, 0, stream>>>(W1, Wt1, GFIN);
    GCN_cvt_wt <<<(64 * GH   + T - 1) / T, T, 0, stream>>>(W2, Wt2, GH);
    GCN_cvt_wt <<<(64 * GH   + T - 1) / T, T, 0, stream>>>(W3, Wt3, GH);
    GCN_cvt_f16<<<gNF, T, 0, stream>>>(x, Xh, GN * GFIN);

    // layer 1
    GCN_gemm_nx64<GFIN><<<gTiles, 128, 0, stream>>>(Xh, Wt1, t);
    GCN_agg_init    <<<gNH,  T, 0, stream>>>(t, dis, b1, agg);
    GCN_edge_scatter<<<gEdg, T, 0, stream>>>(t, src, dst, dis, agg);
    GCN_relu_cvt    <<<gNH,  T, 0, stream>>>(agg, hh);

    // layer 2
    GCN_gemm_nx64<GH><<<gTiles, 128, 0, stream>>>(hh, Wt2, t);
    GCN_agg_init    <<<gNH,  T, 0, stream>>>(t, dis, b2, agg);
    GCN_edge_scatter<<<gEdg, T, 0, stream>>>(t, src, dst, dis, agg);
    GCN_relu_cvt    <<<gNH,  T, 0, stream>>>(agg, hh);

    // layer 3 (no relu)
    GCN_gemm_nx64<GH><<<gTiles, 128, 0, stream>>>(hh, Wt3, t);
    GCN_agg_init    <<<gNH,  T, 0, stream>>>(t, dis, b3, agg);
    GCN_edge_scatter<<<gEdg, T, 0, stream>>>(t, src, dst, dis, agg);

    // pool + classifier
    GCN_zero_pool<<<(GB * GH + T - 1) / T, T, 0, stream>>>(sums, cnt);
    GCN_pool     <<<gNH, T, 0, stream>>>(agg, batch, sums, cnt);
    GCN_logits   <<<(GB * GC + T - 1) / T, T, 0, stream>>>(sums, cnt, Wl, bl, out);
}
